// RegGCNConv_38603166056521
// MI455X (gfx1250) — matrix operand order repaired and checked by reading, not compiled, not run on hardware
//
#include <hip/hip_runtime.h>

typedef float v2f __attribute__((ext_vector_type(2)));
typedef float v4f __attribute__((ext_vector_type(4)));
typedef float v8f __attribute__((ext_vector_type(8)));
typedef unsigned int v4u __attribute__((ext_vector_type(4)));
typedef __bf16 v16bf __attribute__((ext_vector_type(16)));

#define N_VERTEX 16384
#define D 128            // d_in == d_out == 128
#define BS_STRIDE 288    // f32 staging: dwords per k-pair row (256+32 pad)
#define KC 32            // big-GEMM K-chunk (one K=32 slab), double-buffered
#define TST 20           // bf16T staging: dwords per n-row (16 + 4 pad)
#define LDT (N_VERTEX/2) // dwords per n-row of global transposed hi/lo arrays

struct Pair32 { v4u a, b; };   // 32B, bit_cast target for v16bf

__device__ __forceinline__ v4f nt_load4(const float* p) {
    return __builtin_nontemporal_load((const v4f*)p);
}
__device__ __forceinline__ unsigned pack_bf(__bf16 x, __bf16 y) {
    return (unsigned)__builtin_bit_cast(unsigned short, x) |
           ((unsigned)__builtin_bit_cast(unsigned short, y) << 16);
}

// ---------------------------------------------------------------------------
// Elementwise: rwh[i][j] = h[i][j] * reg_param[region_index[i]][j]
// ---------------------------------------------------------------------------
__global__ __launch_bounds__(256) void rwh_kernel(
    const float* __restrict__ h,
    const int*   __restrict__ region_index,
    const float* __restrict__ reg_param,
    float*       __restrict__ rwh)
{
    const int i   = (blockIdx.x * blockDim.x + threadIdx.x) * 4;
    const int row = i >> 7;
    const int col = i & 127;
    const int reg = region_index[row];
    v4f hv = *(const v4f*)(h + i);
    v4f rv = *(const v4f*)(reg_param + reg * D + col);
    v4f o  = hv * rv;
    *(v4f*)(rwh + i) = o;
}

// ---------------------------------------------------------------------------
// Small GEMM (f32 WMMA 16x16x4): C[M][128] = A[M][128] @ B[128][128]
// OMODE 0: write C as f32 [M][128]            (ra)
// OMODE 1: write C split bf16 hi/lo, k-pair packed + transposed:
//          Bh/Bl[n][kp] dword = {bf16(C[2kp][n]), bf16(C[2kp+1][n])}   (tmp)
// ---------------------------------------------------------------------------
template <int OMODE>
__global__ __launch_bounds__(128) void gemm_f32_kernel(
    const float* __restrict__ A,        // [M][128]
    const float* __restrict__ B,        // [128][128]
    float*       __restrict__ outF,     // OMODE 0
    unsigned*    __restrict__ outBh,    // OMODE 1
    unsigned*    __restrict__ outBl)    // OMODE 1
{
    __shared__ float bs[(D / 2) * BS_STRIDE];   // full B, k-pair interleaved

    const int lane   = threadIdx.x & 31;
    const int wave   = threadIdx.x >> 5;
    const int half   = lane >> 4;
    const int l15    = lane & 15;
    const int m_base = blockIdx.x * 64 + wave * 16;
    const size_t a_row = (size_t)(m_base + l15) * D;
    const float* bptr = bs + half * BS_STRIDE + l15 * 2;

    v8f acc[8];
#pragma unroll
    for (int t = 0; t < 8; ++t) acc[t] = {};

    for (int u = threadIdx.x; u < (D / 2) * 64; u += 128) {
        const int kp = u >> 6, n2 = u & 63;
        v2f g0 = *(const v2f*)(B + (size_t)(2 * kp)     * D + 2 * n2);
        v2f g1 = *(const v2f*)(B + (size_t)(2 * kp + 1) * D + 2 * n2);
        v4f st = {g0.x, g1.x, g0.y, g1.y};
        *(v4f*)(bs + kp * BS_STRIDE + 4 * n2) = st;
    }
    __syncthreads();

#pragma unroll 4
    for (int kk = 0; kk < D; kk += 4) {
        v2f a = *(const v2f*)(A + a_row + (size_t)(kk + 2 * half));
        const float* bp = bptr + (kk >> 1) * BS_STRIDE;
        v2f bfrag[8];
#pragma unroll
        for (int t = 0; t < 8; ++t)
            bfrag[t] = *(const v2f*)(bp + t * 32);
#pragma unroll
        for (int t = 0; t < 8; ++t)
            acc[t] = __builtin_amdgcn_wmma_f32_16x16x4_f32(
                false, a, false, bfrag[t], (short)0, acc[t], false, false);
    }

#pragma unroll
    for (int t = 0; t < 8; ++t) {
        const int n = t * 16 + l15;
        if (OMODE == 0) {
#pragma unroll
            for (int r = 0; r < 8; ++r)
                outF[(size_t)(m_base + r + 8 * half) * D + n] = acc[t][r];
        } else {
            const int kp0 = (m_base >> 1) + 4 * half;
            v4u h4, l4;
#pragma unroll
            for (int j = 0; j < 4; ++j) {
                float c0 = acc[t][2 * j], c1 = acc[t][2 * j + 1];
                __bf16 h0 = (__bf16)c0, h1 = (__bf16)c1;
                __bf16 e0 = (__bf16)(c0 - (float)h0);
                __bf16 e1 = (__bf16)(c1 - (float)h1);
                h4[j] = pack_bf(h0, h1);
                l4[j] = pack_bf(e0, e1);
            }
            *(v4u*)(outBh + (size_t)n * LDT + kp0) = h4;
            *(v4u*)(outBl + (size_t)n * LDT + kp0) = l4;
        }
    }
}

// ---------------------------------------------------------------------------
// Big GEMM, error-compensated split-bf16 (Markidis), LDS double-buffered:
//   out = A @ B + addC + bias
//   per K=32 slab & tile: acc += Ah*Bh + Ah*Bl + Al*Bh  (f32 accumulate)
// A f32 streamed NT from HBM, split hi/lo in registers (co-executes w/ WMMA).
// B pre-split/packed/transposed; chunk c+1 prefetched to VGPRs during chunk
// c's compute, stored to the alternate LDS buffer, ONE barrier per chunk.
// ---------------------------------------------------------------------------
__global__ __launch_bounds__(128) void gemm_bf16x3_kernel(
    const float*    __restrict__ A,      // [M][K] f32 adjacency
    int K,
    const unsigned* __restrict__ Bh,     // [128][K/2] packed bf16 pairs
    const unsigned* __restrict__ Bl,
    const float*    __restrict__ addC,   // ra [M][128]
    const float*    __restrict__ bias,   // [128]
    float*          __restrict__ out)    // [M][128]
{
    __shared__ unsigned sh[2][D * TST];   // 2 x 10 KB
    __shared__ unsigned sl[2][D * TST];   // 2 x 10 KB

    const int tid    = threadIdx.x;
    const int lane   = tid & 31;
    const int wave   = tid >> 5;
    const int half   = lane >> 4;
    const int l15    = lane & 15;
    const int m_base = blockIdx.x * 64 + wave * 16;
    const size_t a_row = (size_t)(m_base + l15) * (size_t)K;

    v8f acc[8];
#pragma unroll
    for (int t = 0; t < 8; ++t) acc[t] = {};

    // staging map: i in [0,4): u = i*128+tid -> n = u>>2, j = (u&3)*4
    v4u ph4[4], pl4[4];
#pragma unroll
    for (int i = 0; i < 4; ++i) {
        const int u = i * 128 + tid, n = u >> 2, j = (u & 3) * 4;
        ph4[i] = *(const v4u*)(Bh + (size_t)n * LDT + j);
        pl4[i] = *(const v4u*)(Bl + (size_t)n * LDT + j);
    }
#pragma unroll
    for (int i = 0; i < 4; ++i) {
        const int u = i * 128 + tid, n = u >> 2, j = (u & 3) * 4;
        *(v4u*)(&sh[0][n * TST + j]) = ph4[i];
        *(v4u*)(&sl[0][n * TST + j]) = pl4[i];
    }
    __syncthreads();

    const int NCHUNK = K / KC;
    for (int c = 0; c < NCHUNK; ++c) {
        const int buf = c & 1;

        // --- prefetch chunk c+1 into registers (latency hidden by compute) ---
        if (c + 1 < NCHUNK) {
            const int kp0 = ((c + 1) * KC) >> 1;
#pragma unroll
            for (int i = 0; i < 4; ++i) {
                const int u = i * 128 + tid, n = u >> 2, j = (u & 3) * 4;
                ph4[i] = *(const v4u*)(Bh + (size_t)n * LDT + kp0 + j);
                pl4[i] = *(const v4u*)(Bl + (size_t)n * LDT + kp0 + j);
            }
        }

        // --- compute one K=32 slab from LDS buffer `buf` ---
        const float* ap = A + a_row + (size_t)(c * KC + 16 * half);
        v4f a0 = nt_load4(ap),      a1 = nt_load4(ap + 4);
        v4f a2 = nt_load4(ap + 8),  a3 = nt_load4(ap + 12);
        float af[16];
        *(v4f*)(af + 0) = a0; *(v4f*)(af + 4)  = a1;
        *(v4f*)(af + 8) = a2; *(v4f*)(af + 12) = a3;
        v16bf ah, al;
#pragma unroll
        for (int i = 0; i < 16; ++i) {
            __bf16 hb = (__bf16)af[i];
            ah[i] = hb;
            al[i] = (__bf16)(af[i] - (float)hb);
        }

        const unsigned* phb = &sh[buf][l15 * TST + 8 * half];
        const unsigned* plb = &sl[buf][l15 * TST + 8 * half];
#pragma unroll
        for (int g = 0; g < 2; ++g) {
            v16bf bhf[4], blf[4];
#pragma unroll
            for (int t = 0; t < 4; ++t) {
                const int off = (g * 4 + t) * 16 * TST;
                Pair32 p;
                p.a = *(const v4u*)(phb + off);
                p.b = *(const v4u*)(phb + off + 4);
                bhf[t] = __builtin_bit_cast(v16bf, p);
                p.a = *(const v4u*)(plb + off);
                p.b = *(const v4u*)(plb + off + 4);
                blf[t] = __builtin_bit_cast(v16bf, p);
            }
#pragma unroll
            for (int t = 0; t < 4; ++t)
                acc[g * 4 + t] = __builtin_amdgcn_wmma_f32_16x16x32_bf16(
                    false, ah, false, bhf[t], (short)0, acc[g * 4 + t],
                    false, false);
#pragma unroll
            for (int t = 0; t < 4; ++t)
                acc[g * 4 + t] = __builtin_amdgcn_wmma_f32_16x16x32_bf16(
                    false, ah, false, blf[t], (short)0, acc[g * 4 + t],
                    false, false);
#pragma unroll
            for (int t = 0; t < 4; ++t)
                acc[g * 4 + t] = __builtin_amdgcn_wmma_f32_16x16x32_bf16(
                    false, al, false, bhf[t], (short)0, acc[g * 4 + t],
                    false, false);
        }

        // --- store prefetched chunk to the alternate buffer ---
        if (c + 1 < NCHUNK) {
#pragma unroll
            for (int i = 0; i < 4; ++i) {
                const int u = i * 128 + tid, n = u >> 2, j = (u & 3) * 4;
                *(v4u*)(&sh[buf ^ 1][n * TST + j]) = ph4[i];
                *(v4u*)(&sl[buf ^ 1][n * TST + j]) = pl4[i];
            }
        }
        __syncthreads();   // single barrier per chunk (classic double-buffer)
    }

    // --- epilogue: acc + ra + bias -> out ---
#pragma unroll
    for (int t = 0; t < 8; ++t) {
        const int n = t * 16 + l15;
        const float bn = bias[n];
#pragma unroll
        for (int r = 0; r < 8; ++r) {
            const int row = m_base + r + 8 * half;
            out[(size_t)row * D + n] = acc[t][r] + addC[(size_t)row * D + n] + bn;
        }
    }
}

// ---------------------------------------------------------------------------
// Launch:
//   rwh      = h * reg_param[region_index]
//   ra       = rwh @ weight                       (f32 WMMA)
//   tmpBh/Bl = split(rwh @ lag_weight)            (f32 WMMA, packed-T output)
//   out      = norm_adj @ tmp + ra + bias         (split-bf16 WMMA, fused)
// ---------------------------------------------------------------------------
extern "C" void kernel_launch(void* const* d_in, const int* in_sizes, int n_in,
                              void* d_out, int out_size, void* d_ws, size_t ws_size,
                              hipStream_t stream) {
    const float* h    = (const float*)d_in[0];
    const float* adj  = (const float*)d_in[1];
    const int*   ridx = (const int*)  d_in[2];
    const float* w    = (const float*)d_in[3];
    const float* lagw = (const float*)d_in[4];
    const float* regp = (const float*)d_in[5];
    const float* bias = (const float*)d_in[6];
    float* out = (float*)d_out;

    float*    rwh = (float*)d_ws;                           // 8 MB
    float*    ra  = rwh + (size_t)N_VERTEX * D;             // 8 MB
    unsigned* tbh = (unsigned*)(ra + (size_t)N_VERTEX * D); // 4 MB
    unsigned* tbl = tbh + (size_t)D * LDT;                  // 4 MB

    rwh_kernel<<<(N_VERTEX * D) / (256 * 4), 256, 0, stream>>>(h, ridx, regp, rwh);

    dim3 grid(N_VERTEX / 64), block(128);
    gemm_f32_kernel<0><<<grid, block, 0, stream>>>(rwh, w,    ra,      nullptr, nullptr);
    gemm_f32_kernel<1><<<grid, block, 0, stream>>>(rwh, lagw, nullptr, tbh,     tbl);
    gemm_bf16x3_kernel<<<grid, block, 0, stream>>>(adj, N_VERTEX, tbh, tbl, ra, bias, out);
}
